// CriticalSparseAttention_24283745092138
// MI455X (gfx1250) — compile-verified, matrix operand order checked by reading, not run on hardware
//
#include <hip/hip_runtime.h>
#include <math.h>

#define DEVFN __device__ __forceinline__

typedef __attribute__((ext_vector_type(16))) _Float16 v16h;
typedef __attribute__((ext_vector_type(8)))  _Float16 v8h;
typedef __attribute__((ext_vector_type(8)))  float    v8f;
typedef __attribute__((ext_vector_type(4)))  float    v4f;

static constexpr int    BB   = 8;
static constexpr int    LL   = 4096;
static constexpr int    HH   = 512;
static constexpr int    CC   = 64;
static constexpr int    AA   = 512;
static constexpr float  NEGV = -1000000000.0f;

// ---------------------------------------------------------------------------
// helpers
// ---------------------------------------------------------------------------
DEVFN v8f wmma16(v16h a, v16h b, v8f c) {
  // v_wmma_f32_16x16x32_f16
  return __builtin_amdgcn_wmma_f32_16x16x32_f16(false, a, false, b, (short)0, c,
                                                false, false);
}

DEVFN v16h cat16(v8h lo, v8h hi) {
  return __builtin_shufflevector(lo, hi, 0, 1, 2, 3, 4, 5, 6, 7, 8, 9, 10, 11,
                                 12, 13, 14, 15);
}

DEVFN v8h cvt8(const float* p) {
  v4f x = *reinterpret_cast<const v4f*>(p);
  v4f y = *reinterpret_cast<const v4f*>(p + 4);
  v8h r;
  r[0] = (_Float16)x[0]; r[1] = (_Float16)x[1];
  r[2] = (_Float16)x[2]; r[3] = (_Float16)x[3];
  r[4] = (_Float16)y[0]; r[5] = (_Float16)y[1];
  r[6] = (_Float16)y[2]; r[7] = (_Float16)y[3];
  return r;
}

DEVFN v8h cvt8_relu_scale(const float* p, float s) {
  v4f x = *reinterpret_cast<const v4f*>(p);
  v4f y = *reinterpret_cast<const v4f*>(p + 4);
  v8h r;
  r[0] = (_Float16)(fmaxf(x[0], 0.f) * s); r[1] = (_Float16)(fmaxf(x[1], 0.f) * s);
  r[2] = (_Float16)(fmaxf(x[2], 0.f) * s); r[3] = (_Float16)(fmaxf(x[3], 0.f) * s);
  r[4] = (_Float16)(fmaxf(y[0], 0.f) * s); r[5] = (_Float16)(fmaxf(y[1], 0.f) * s);
  r[6] = (_Float16)(fmaxf(y[2], 0.f) * s); r[7] = (_Float16)(fmaxf(y[3], 0.f) * s);
  return r;
}

// K-offset within a 32-wide K tile for the 16-bit A/B WMMA fragment layout:
// lanes 0-15 hold K in {0..7, 16..23}; lanes 16-31 hold K in {8..15, 24..31}.
DEVFN int kidx(int lane, int i) {
  return ((lane >> 4) << 3) + (i & 7) + ((i >> 3) << 4);
}

// ---------------------------------------------------------------------------
// prep kernels
// ---------------------------------------------------------------------------
__global__ void kInit(int* cmax) {
  if (threadIdx.x < BB) cmax[threadIdx.x] = 0;
}

// hidden f32 -> f16 row-major (A-fragment friendly: K contiguous per row)
__global__ void kHidF16(const float* __restrict__ h, _Float16* __restrict__ o) {
  int idx = blockIdx.x * 256 + threadIdx.x;
  o[idx] = (_Float16)h[idx];
}

// hidden f32 -> f16 swizzled into WMMA-B fragment order for GEMM2
// (K = l over 128 tiles of 32, N = h over 32 tiles of 16).
// out layout: [b][kt][nt][lane][i]  (16 halfs contiguous per lane)
// Two-phase: coalesced 32x512 f32 tile load -> LDS (padded) -> coalesced
// swizzled f16 store.
__global__ void kHidSwz(const float* __restrict__ h, _Float16* __restrict__ o) {
  __shared__ _Float16 tile[32 * 520];          // padded stride: bank-conflict fix
  int bkt = blockIdx.x;                        // b*128 + kt
  int b   = bkt >> 7;
  int kt  = bkt & 127;
  const float* src = h + ((size_t)b * LL + (size_t)kt * 32) * HH;
#pragma unroll
  for (int j = 0; j < 64; ++j) {
    int e = threadIdx.x + j * 256;             // 0..16383 over 32x512
    tile[(e >> 9) * 520 + (e & 511)] = (_Float16)src[e];
  }
  __syncthreads();
  _Float16* dst = o + (size_t)bkt * (32 * 512);
#pragma unroll
  for (int j = 0; j < 64; ++j) {
    int e    = threadIdx.x + j * 256;
    int i    = e & 15;
    int lane = (e >> 4) & 31;
    int nt   = e >> 9;
    int l32  = kidx(lane, i);
    int hc   = nt * 16 + (lane & 15);
    dst[e]   = tile[l32 * 520 + hc];
  }
}

// proj_w (512x512 row-major, K=H rows, N=A cols) -> WMMA-B fragment order
// out layout: [nt(32)][kt(16)][lane(32)][i(16)]  -- same two-phase transpose
__global__ void kProjSwz(const float* __restrict__ w, _Float16* __restrict__ o) {
  __shared__ _Float16 tile[32 * 520];
  int kt = blockIdx.x;                         // 0..15
  const float* src = w + (size_t)kt * 32 * AA;
#pragma unroll
  for (int j = 0; j < 64; ++j) {
    int e = threadIdx.x + j * 256;             // 32x512 tile (k32 x n)
    tile[(e >> 9) * 520 + (e & 511)] = (_Float16)src[e];
  }
  __syncthreads();
#pragma unroll
  for (int j = 0; j < 64; ++j) {
    int e    = threadIdx.x + j * 256;
    int i    = e & 15;
    int lane = (e >> 4) & 31;
    int nt   = e >> 9;
    int k32  = kidx(lane, i);
    int n    = nt * 16 + (lane & 15);
    o[((size_t)nt * 16 + kt) * 512 + (e & 511)] = tile[k32 * 520 + n];
  }
}

// per (b,l): relu(activity) -> transposed (B,C,L), row sums, count max per b
__global__ void kActPrep(const float* __restrict__ chan, const int* __restrict__ cnt,
                         float* __restrict__ actT, float* __restrict__ sumAct,
                         int* __restrict__ cmax) {
  int idx = blockIdx.x * 256 + threadIdx.x;   // (b,l)
  int b   = idx >> 12;
  int l   = idx & 4095;
  const float* p = chan + (size_t)idx * CC;
  float s = 0.f;
  for (int c = 0; c < CC; ++c) {
    float a = fmaxf(p[c], 0.f);
    s += a;
    actT[((size_t)b * CC + c) * LL + l] = a;
  }
  sumAct[idx] = s;
  atomicMax(&cmax[b], cnt[idx]);
}

// ---------------------------------------------------------------------------
// GEMM1: logits = tanh(hidden @ proj_w + proj_b) @ score_w + score_b
//        + log1p(max(count,0))   -- fused, WMMA f16, K=512
// One wave = 16 rows.  A fragments are staged in per-wave LDS (16 KB/wave).
// A __syncthreads() makes the staging opaque (no store-to-load forwarding),
// and the k-tile order is rotated by nt so no fragment load is loop-invariant
// -> LICM cannot hoist the 16-fragment array into registers -> no spills.
// ---------------------------------------------------------------------------
__global__ void kLogits(const _Float16* __restrict__ hidf16,
                        const _Float16* __restrict__ bswz,
                        const float* __restrict__ projb,
                        const float* __restrict__ scorew,
                        const float* __restrict__ scoreb,
                        const int* __restrict__ cnt,
                        float* __restrict__ logits) {
  __shared__ _Float16 alds[8 * 16 * 32 * 16];   // 8 waves x 16 frags x 1 KB
  int lane    = threadIdx.x & 31;
  int wave    = threadIdx.x >> 5;
  int rowbase = (blockIdx.x * 8 + wave) * 16;
  int m       = lane & 15;
  int kbase   = (lane >> 4) * 8;
  int abase   = wave * (16 * 32 * 16);

  const _Float16* hp = hidf16 + ((size_t)rowbase + m) * HH;
#pragma unroll
  for (int kt = 0; kt < 16; ++kt) {
    v8h lo = *reinterpret_cast<const v8h*>(hp + kt * 32 + kbase);
    v8h hi = *reinterpret_cast<const v8h*>(hp + kt * 32 + kbase + 16);
    *reinterpret_cast<v16h*>(&alds[abase + (kt * 32 + lane) * 16]) = cat16(lo, hi);
  }
  __syncthreads();   // barrier: forces real ds_load of the fragments below

  float sacc[8];
#pragma unroll
  for (int r = 0; r < 8; ++r) sacc[r] = 0.f;

  for (int nt = 0; nt < 32; ++nt) {
    v8f c = {};
    // rotate k-tile order by nt: same fragment set per iteration, but no
    // individual load address is nt-invariant (defeats LICM hoist+spill)
#pragma unroll
    for (int kt = 0; kt < 16; ++kt) {
      int  ktp = (kt + nt) & 15;
      v16h af  = *reinterpret_cast<const v16h*>(
          &alds[abase + (ktp * 32 + lane) * 16]);
      v16h bf  = *reinterpret_cast<const v16h*>(
          bswz + (((size_t)nt * 16 + ktp) * 32 + lane) * 16);
      c = wmma16(af, bf, c);
    }
    float pb = projb[nt * 16 + (lane & 15)];
    float sw = scorew[nt * 16 + (lane & 15)];
#pragma unroll
    for (int r = 0; r < 8; ++r) sacc[r] += tanhf(c[r] + pb) * sw;
  }

  // reduce over N (the 16 lanes of each half-wave)
#pragma unroll
  for (int off = 8; off; off >>= 1) {
#pragma unroll
    for (int r = 0; r < 8; ++r) sacc[r] += __shfl_xor(sacc[r], off, 16);
  }

  if ((lane & 15) == 0) {
    int half = lane >> 4;
#pragma unroll
    for (int r = 0; r < 8; ++r) {
      int   row = rowbase + r + half * 8;
      float cv  = (float)cnt[row];
      logits[row] = sacc[r] + scoreb[0] + log1pf(fmaxf(cv, 0.f));
    }
  }
}

// ---------------------------------------------------------------------------
// softmax over L per (b,c); writes weights output; records has_any
// ---------------------------------------------------------------------------
__global__ void kSoftmax(const float* __restrict__ actT,
                         const float* __restrict__ logits,
                         float* __restrict__ wout, int* __restrict__ hasAny) {
  __shared__ float red[8];
  int bc = blockIdx.x;               // b*64 + c
  int b  = bc >> 6;
  const float* ap = actT + (size_t)bc * LL;
  const float* lp = logits + (size_t)b * LL;

  float v[16];
  float mx = -3.0e38f;
#pragma unroll
  for (int j = 0; j < 16; ++j) {
    int   l   = threadIdx.x + j * 256;
    float act = ap[l];
    float val = (act > 0.f) ? (lp[l] + log1pf(act)) : NEGV;
    v[j] = val;
    mx   = fmaxf(mx, val);
  }
#pragma unroll
  for (int off = 16; off; off >>= 1) mx = fmaxf(mx, __shfl_xor(mx, off, 32));
  if ((threadIdx.x & 31) == 0) red[threadIdx.x >> 5] = mx;
  __syncthreads();
  float bm = red[0];
#pragma unroll
  for (int w = 1; w < 8; ++w) bm = fmaxf(bm, red[w]);
  __syncthreads();

  float s = 0.f;
#pragma unroll
  for (int j = 0; j < 16; ++j) s += __expf(v[j] - bm);
#pragma unroll
  for (int off = 16; off; off >>= 1) s += __shfl_xor(s, off, 32);
  if ((threadIdx.x & 31) == 0) red[threadIdx.x >> 5] = s;
  __syncthreads();
  float bs = 0.f;
#pragma unroll
  for (int w = 0; w < 8; ++w) bs += red[w];

  int   any = (bm > -1.0e8f) ? 1 : 0;
  float inv = any ? (1.f / bs) : 0.f;
#pragma unroll
  for (int j = 0; j < 16; ++j) {
    int l = threadIdx.x + j * 256;
    wout[(size_t)bc * LL + l] = __expf(v[j] - bm) * inv;
  }
  if (threadIdx.x == 0) hasAny[bc] = any;
}

// ---------------------------------------------------------------------------
// GEMM2: ctx(b,c,h) = sum_l weights(b,c,l) * hidden(b,l,h)   K=4096
// ---------------------------------------------------------------------------
__global__ void kCtx(const float* __restrict__ weights,
                     const _Float16* __restrict__ hidswz,
                     float* __restrict__ ctx) {
  int b     = blockIdx.y;
  int wave  = threadIdx.x >> 5;
  int lane  = threadIdx.x & 31;
  int mt    = wave & 3;                      // c tile (4 tiles of 16 -> C=64)
  int nt    = blockIdx.x * 2 + (wave >> 2);  // h tile 0..31
  int ch    = mt * 16 + (lane & 15);
  int kbase = (lane >> 4) * 8;

  const float* wp = weights + ((size_t)b * CC + ch) * LL;
  v8f acc = {};
#pragma unroll 4
  for (int kt = 0; kt < 128; ++kt) {
    v8h  lo = cvt8(wp + kt * 32 + kbase);
    v8h  hi = cvt8(wp + kt * 32 + kbase + 16);
    v16h af = cat16(lo, hi);
    v16h bf = *reinterpret_cast<const v16h*>(
        hidswz + ((((size_t)b * 128 + kt) * 32 + nt) * 32 + lane) * 16);
    acc = wmma16(af, bf, acc);
  }
  int h = nt * 16 + (lane & 15);
#pragma unroll
  for (int r = 0; r < 8; ++r) {
    int cc = mt * 16 + r + 8 * (lane >> 4);
    ctx[((size_t)b * CC + cc) * HH + h] = acc[r];
  }
}

// ctx -> f16 WMMA-B order (K=c: 2 tiles of 32, N=h: 32 tiles of 16),
// applying the no-active-channel fallback (hidden[b, L-1, :]).
// layout: [b][kt(2)][nt(32)][lane(32)][i(16)]
__global__ void kCtxSwz(const float* __restrict__ ctx, const int* __restrict__ hasAny,
                        const float* __restrict__ hidden, _Float16* __restrict__ o) {
  int idx  = blockIdx.x * 256 + threadIdx.x;
  int i    = idx & 15;
  int lane = (idx >> 4) & 31;
  int nt   = (idx >> 9) & 31;
  int kt   = (idx >> 14) & 1;
  int b    = idx >> 15;
  int c    = kt * 32 + kidx(lane, i);
  int h    = nt * 16 + (lane & 15);
  float v  = hasAny[b * CC + c]
                 ? ctx[((size_t)b * CC + c) * HH + h]
                 : hidden[((size_t)b * LL + (LL - 1)) * HH + h];
  o[idx] = (_Float16)v;
}

// ---------------------------------------------------------------------------
// GEMM3 + epilogue: attended = where(mask, (mix @ ctx) * gate, hidden)  K=64
// ---------------------------------------------------------------------------
__global__ void kExpand(const float* __restrict__ chan,
                        const float* __restrict__ sumAct,
                        const _Float16* __restrict__ ctxswz,
                        const unsigned char* __restrict__ cmask,
                        const int* __restrict__ cnt, const int* __restrict__ cmax,
                        const float* __restrict__ hidden,
                        float* __restrict__ outAtt) {
  int b     = blockIdx.y;
  int lbase = blockIdx.x * 128 + (threadIdx.x >> 5) * 16;
  int lane  = threadIdx.x & 31;
  int kbase = (lane >> 4) * 8;

  // A fragments: mix = relu(activity)/max(rowsum,1), rows = l, K = c (64)
  size_t rowoff = (size_t)b * LL + lbase + (lane & 15);
  float  inv    = 1.f / fmaxf(sumAct[rowoff], 1.f);
  const float* ap = chan + rowoff * CC;
  v16h a[2];
#pragma unroll
  for (int kt = 0; kt < 2; ++kt) {
    v8h lo = cvt8_relu_scale(ap + kt * 32 + kbase, inv);
    v8h hi = cvt8_relu_scale(ap + kt * 32 + kbase + 16, inv);
    a[kt]  = cat16(lo, hi);
  }

  float maxcf = fmaxf((float)cmax[b], 1.f);
  float gate[8];
  int   msk[8];
#pragma unroll
  for (int r = 0; r < 8; ++r) {
    int    lr = lbase + r + 8 * (lane >> 4);
    size_t ro = (size_t)b * LL + lr;
    gate[r]   = 1.f + (float)cnt[ro] / maxcf;
    msk[r]    = cmask[ro];
  }

  for (int nt = 0; nt < 32; ++nt) {
    v8f acc = {};
#pragma unroll
    for (int kt = 0; kt < 2; ++kt) {
      v16h bf = *reinterpret_cast<const v16h*>(
          ctxswz + ((((size_t)b * 2 + kt) * 32 + nt) * 32 + lane) * 16);
      acc = wmma16(a[kt], bf, acc);
    }
    int h = nt * 16 + (lane & 15);
#pragma unroll
    for (int r = 0; r < 8; ++r) {
      int    lr  = lbase + r + 8 * (lane >> 4);
      size_t off = ((size_t)b * LL + lr) * HH + h;
      outAtt[off] = msk[r] ? acc[r] * gate[r] : hidden[off];
    }
  }
}

// ---------------------------------------------------------------------------
// launch
// ---------------------------------------------------------------------------
extern "C" void kernel_launch(void* const* d_in, const int* in_sizes, int n_in,
                              void* d_out, int out_size, void* d_ws, size_t ws_size,
                              hipStream_t stream) {
  (void)in_sizes; (void)n_in; (void)out_size; (void)ws_size;

  const float*         hidden = (const float*)d_in[0];
  const unsigned char* cmask  = (const unsigned char*)d_in[1];
  const int*           cnt    = (const int*)d_in[2];
  const float*         chan   = (const float*)d_in[3];
  const float*         projw  = (const float*)d_in[4];
  const float*         projb  = (const float*)d_in[5];
  const float*         scorew = (const float*)d_in[6];
  const float*         scoreb = (const float*)d_in[7];

  float* outAtt = (float*)d_out;                                // (B,L,H)
  float* outW   = outAtt + (size_t)BB * LL * HH;                // (B,C,L)

  char* ws = (char*)d_ws;
  size_t o = 0;
  _Float16* hidf16 = (_Float16*)(ws + o); o += (size_t)BB * LL * HH * 2;  // 32 MB
  _Float16* hidswz = (_Float16*)(ws + o); o += (size_t)BB * LL * HH * 2;  // 32 MB
  _Float16* bswz   = (_Float16*)(ws + o); o += (size_t)HH * AA * 2;       // 512 KB
  _Float16* ctxswz = (_Float16*)(ws + o); o += (size_t)BB * CC * HH * 2;  // 512 KB
  float*    logits = (float*)(ws + o);    o += (size_t)BB * LL * 4;       // 128 KB
  float*    sumAct = (float*)(ws + o);    o += (size_t)BB * LL * 4;       // 128 KB
  float*    actT   = (float*)(ws + o);    o += (size_t)BB * CC * LL * 4;  // 8 MB
  float*    ctx    = (float*)(ws + o);    o += (size_t)BB * CC * HH * 4;  // 1 MB
  int*      hasAny = (int*)(ws + o);      o += (size_t)BB * CC * 4;
  int*      cmax   = (int*)(ws + o);      o += 256;

  hipLaunchKernelGGL(kInit, dim3(1), dim3(32), 0, stream, cmax);
  hipLaunchKernelGGL(kHidF16, dim3((BB * LL * HH) / 256), dim3(256), 0, stream,
                     hidden, hidf16);
  hipLaunchKernelGGL(kHidSwz, dim3(BB * 128), dim3(256), 0, stream,
                     hidden, hidswz);
  hipLaunchKernelGGL(kProjSwz, dim3(16), dim3(256), 0, stream,
                     projw, bswz);
  hipLaunchKernelGGL(kActPrep, dim3((BB * LL) / 256), dim3(256), 0, stream,
                     chan, cnt, actT, sumAct, cmax);
  hipLaunchKernelGGL(kLogits, dim3((BB * LL) / 128), dim3(256), 0, stream,
                     hidf16, bswz, projb, scorew, scoreb, cnt, logits);
  hipLaunchKernelGGL(kSoftmax, dim3(BB * CC), dim3(256), 0, stream,
                     actT, logits, outW, hasAny);
  hipLaunchKernelGGL(kCtx, dim3(16, BB), dim3(256), 0, stream,
                     outW, hidswz, ctx);
  hipLaunchKernelGGL(kCtxSwz, dim3((BB * CC * HH) / 256), dim3(256), 0, stream,
                     ctx, hasAny, hidden, ctxswz);
  hipLaunchKernelGGL(kExpand, dim3(LL / 128, BB), dim3(256), 0, stream,
                     chan, sumAct, ctxswz, cmask, cnt, cmax, hidden, outAtt);
}
